// UnetGraphSAGE_11278584119663
// MI455X (gfx1250) — matrix operand order repaired and checked by reading, not compiled, not run on hardware
//
#include <hip/hip_runtime.h>
#include <hip/hip_bf16.h>
#include <math.h>

// ============================================================================
// UNet-GraphSAGE forward for MI455X (gfx1250).
// Dense linear algebra -> wave32 WMMA (v_wmma_f32_16x16x32_f16, f32 accum) on
// zero-padded f16 operands so the inner loop is pure b128 loads + wmma (no
// predicated scalar loads). NNConv refactored: msg = Q[src] + g[e].P[src],
// P = h @ Wp (node-level GEMM): 4x fewer FLOPs than per-edge HxH weights
// (E = 4N) and avoids a 402 MB We tensor on the finest graph.
// ============================================================================

typedef __attribute__((ext_vector_type(16))) _Float16 v16h;
typedef __attribute__((ext_vector_type(8)))  _Float16 v8h;
typedef __attribute__((ext_vector_type(8)))  float    v8f;

#define EHID 32  // edge-MLP hidden width (EH in reference)

static __device__ __forceinline__ float sigmoidf_(float v) {
  return 1.0f / (1.0f + __expf(-v));
}

// ---------------- pack: f32 (M x K) -> zero-padded f16 (Mp x Kp) ------------
__global__ void k_pack_f16(const float* __restrict__ in, _Float16* __restrict__ out,
                           int M, int K, int Mp, int Kp) {
  long idx = (long)blockIdx.x * blockDim.x + threadIdx.x;
  long total = (long)Mp * Kp;
  if (idx >= total) return;
  int k = (int)(idx % Kp);
  long m = idx / Kp;
  out[idx] = (m < M && k < K) ? (_Float16)in[m * K + k] : (_Float16)0.0f;
}

// ---------------- WMMA GEMM on padded f16: C = act(A @ W^T + bias) ----------
// A: Mp x Kp f16, W: Np x Kp f16 (both zero-padded), C: M x N f32.
// One 16x16 tile per wave, 4 waves per block. Inner loop: 4x b128 load + wmma.
__global__ void k_gemm16(const _Float16* __restrict__ A, const _Float16* __restrict__ W,
                         const float* __restrict__ bias, float* __restrict__ C,
                         int M, int N, int Kp, int relu)
{
  const int Np = (N + 15) & ~15;
  const int lane = threadIdx.x;                       // 0..31 (wave32)
  const int row0 = blockIdx.y * 16;
  const int col0 = (blockIdx.x * 4 + threadIdx.y) * 16;
  if (col0 >= Np) return;                             // wave-uniform
  const int mr = lane & 15;
  const int hi = lane >> 4;
  const _Float16* Arow = A + (size_t)(row0 + mr) * Kp;
  const _Float16* Wrow = W + (size_t)(col0 + mr) * Kp;
  v8f acc = {};
  for (int kb = 0; kb < Kp; kb += 32) {
    // A fragment (16x32): lane holds row, halves [kb+hi*8, +8) and +16
    const v8h* ap = (const v8h*)(Arow + kb + hi * 8);
    v8h a0 = ap[0];
    v8h a1 = ap[2];                                   // +16 halves
    v16h a = __builtin_shufflevector(a0, a1, 0, 1, 2, 3, 4, 5, 6, 7,
                                     8, 9, 10, 11, 12, 13, 14, 15);
    // B fragment (32x16): B[k][n] = W[n][k]; same k order as A
    const v8h* bp = (const v8h*)(Wrow + kb + hi * 8);
    v16h b = __builtin_shufflevector(bp[0], bp[2], 0, 1, 2, 3, 4, 5, 6, 7,
                                     8, 9, 10, 11, 12, 13, 14, 15);
    acc = __builtin_amdgcn_wmma_f32_16x16x32_f16(false, a, false, b,
                                                 (short)0, acc, false, false);
  }
#pragma unroll
  for (int j = 0; j < 8; ++j) {
    int r = row0 + hi * 8 + j;
    int c = col0 + mr;
    if (r < M && c < N) {
      float v = acc[j] + (bias ? bias[c] : 0.0f);
      if (relu) v = fmaxf(v, 0.0f);
      C[(size_t)r * N + c] = v;
    }
  }
}

// ---------------- helpers ---------------------------------------------------
__global__ void k_zero(float* __restrict__ p, long n) {
  long i = (long)blockIdx.x * blockDim.x + threadIdx.x;
  if (i < n) p[i] = 0.0f;
}

__global__ void k_deg(const int* __restrict__ dst, float* __restrict__ deg, int E) {
  int e = blockIdx.x * blockDim.x + threadIdx.x;
  if (e < E) atomicAdd(&deg[dst[e]], 1.0f);
}

// Wp[(k*H+o)*H + i] = ew2[(i*H+o)*EHID + k]; Wq[o*H + i] = eb2[i*H+o]
__global__ void k_reorder_ew2(const float* __restrict__ ew2,
                              const float* __restrict__ eb2,
                              float* __restrict__ Wp, float* __restrict__ Wq, int H) {
  int idx = blockIdx.x * blockDim.x + threadIdx.x;
  int total = EHID * H * H;
  if (idx >= total) return;
  int i = idx % H;
  int o = (idx / H) % H;
  int k = idx / (H * H);
  Wp[(size_t)(k * H + o) * H + i] = ew2[(size_t)(i * H + o) * EHID + k];
  if (k == 0) Wq[(size_t)o * H + i] = eb2[(size_t)i * H + o];
}

// msg[e,o] = Q[src,o] + sum_k g[e,k] * P[src, k*H+o];  atomically scatter to acc[dst]
__global__ void k_edge_msg_scatter(const float* __restrict__ g,
                                   const float* __restrict__ P,
                                   const float* __restrict__ Q,
                                   const int* __restrict__ src,
                                   const int* __restrict__ dst,
                                   float* __restrict__ acc, int E, int H) {
  long idx = (long)blockIdx.x * blockDim.x + threadIdx.x;
  if (idx >= (long)E * H) return;
  int e = (int)(idx / H);
  int o = (int)(idx % H);
  int s = src[e];
  const float* Pp = P + (size_t)s * EHID * H + o;
  const float* gp = g + (size_t)e * EHID;
  float m = Q[(size_t)s * H + o];
#pragma unroll 8
  for (int k = 0; k < EHID; ++k) m += gp[k] * Pp[(size_t)k * H];
  atomicAdd(&acc[(size_t)dst[e] * H + o], m);
}

// x = relu(acc / max(deg,1) + cb)
__global__ void k_agg_relu(const float* __restrict__ acc, const float* __restrict__ deg,
                           const float* __restrict__ cb, float* __restrict__ x,
                           int N, int H) {
  long idx = (long)blockIdx.x * blockDim.x + threadIdx.x;
  if (idx >= (long)N * H) return;
  int n = (int)(idx / H);
  int o = (int)(idx % H);
  float d = fmaxf(deg[n], 1.0f);
  x[idx] = fmaxf(acc[idx] / d + cb[o], 0.0f);
}

// single-step GRU gating
__global__ void k_gru(const float* __restrict__ gi, const float* __restrict__ gh,
                      const float* __restrict__ hid, float* __restrict__ hout,
                      int N, int H) {
  long idx = (long)blockIdx.x * blockDim.x + threadIdx.x;
  if (idx >= (long)N * H) return;
  int n = (int)(idx / H);
  int j = (int)(idx % H);
  const float* gin = gi + (size_t)n * 3 * H;
  const float* ghn = gh + (size_t)n * 3 * H;
  float r  = sigmoidf_(gin[j]         + ghn[j]);
  float z  = sigmoidf_(gin[H + j]     + ghn[H + j]);
  float nn = tanhf(gin[2 * H + j] + r * ghn[2 * H + j]);
  hout[idx] = (1.0f - z) * nn + z * hid[idx];
}

// AvgPool2d(2,2) on node-major (6*Hh*Ww, C) -> (6*(Hh/2)*(Ww/2), C)
__global__ void k_pool2(const float* __restrict__ in, float* __restrict__ out,
                        int Hh, int Ww, int C) {
  int Ho = Hh / 2, Wo = Ww / 2;
  long total = 6L * Ho * Wo * C;
  long idx = (long)blockIdx.x * blockDim.x + threadIdx.x;
  if (idx >= total) return;
  int c = (int)(idx % C);
  long node = idx / C;
  int x = (int)(node % Wo);
  int y = (int)((node / Wo) % Ho);
  int f = (int)(node / ((long)Wo * Ho));
  long base = ((long)(f * Hh + 2 * y)) * Ww + 2 * x;
  float s = in[base * C + c] + in[(base + 1) * C + c] +
            in[(base + Ww) * C + c] + in[(base + Ww + 1) * C + c];
  out[idx] = 0.25f * s;
}

// reference's scrambled reshape on h3->h4 pool: out(96,64); t=a*96+b,
// out[b,a] = tmp[(t%384)*16 + t/384]  (tmp: (6*8*8 nodes, 16 pseudo-ch))
__global__ void k_scramble_h4(const float* __restrict__ tmp, float* __restrict__ out) {
  int idx = blockIdx.x * blockDim.x + threadIdx.x;
  if (idx >= 96 * 64) return;
  int b = idx / 64, a = idx % 64;
  int t = a * 96 + b;
  int c = t / 384, node = t % 384;
  out[idx] = tmp[node * 16 + c];
}

// ConvTranspose2d(k=2, s=2): in (6*Hh*Ww, Cin) node-major -> (6*2Hh*2Ww, Cout)
// w: (Cin, Cout, 2, 2)
__global__ void k_convT(const float* __restrict__ in, const float* __restrict__ w,
                        const float* __restrict__ b, float* __restrict__ out,
                        int Hh, int Ww, int Cin, int Cout) {
  int Ho = 2 * Hh, Wo = 2 * Ww;
  long total = 6L * Ho * Wo * Cout;
  long idx = (long)blockIdx.x * blockDim.x + threadIdx.x;
  if (idx >= total) return;
  int o = (int)(idx % Cout);
  long node = idx / Cout;
  int xx = (int)(node % Wo);
  int yy = (int)((node / Wo) % Ho);
  int f  = (int)(node / ((long)Wo * Ho));
  int y = yy >> 1, x = xx >> 1, a = yy & 1, bb = xx & 1;
  const float* ip = in + ((size_t)(f * Hh + y) * Ww + x) * Cin;
  float s = b[o];
  for (int c = 0; c < Cin; ++c)
    s += ip[c] * w[((size_t)(c * Cout + o) * 2 + a) * 2 + bb];
  out[idx] = s;
}

__global__ void k_concat(const float* __restrict__ A, const float* __restrict__ B,
                         float* __restrict__ out, int N, int Ca, int Cb) {
  int Ct = Ca + Cb;
  long idx = (long)blockIdx.x * blockDim.x + threadIdx.x;
  if (idx >= (long)N * Ct) return;
  int c = (int)(idx % Ct);
  long n = idx / Ct;
  out[idx] = (c < Ca) ? A[n * Ca + c] : B[n * Cb + (c - Ca)];
}

// ---------------- host-side orchestration -----------------------------------
namespace {

struct Arena {
  char* base; size_t cap; size_t off;
  void* raw(size_t bytes) {
    bytes = (bytes + 255) & ~(size_t)255;
    void* p = (void*)(base + off);
    off += bytes;
    return p;
  }
  float* f(size_t n) { return (float*)raw(n * sizeof(float)); }
  _Float16* h(size_t n) { return (_Float16*)raw(n * sizeof(_Float16)); }
};

struct MPW {
  const float *pw1, *pb1, *pw2, *pb2;
  const float *ew1, *eb1, *ew2, *eb2, *cb;
  const float *gwih, *gbih, *gwhh, *gbhh;
  const float *dw1, *db1, *dw2, *db2;
};

struct GemmCtx {
  hipStream_t s;
  _Float16* Ah;   // staging for packed A (max 4M halves)
  _Float16* Wh;   // staging for packed W (max 1M halves)
};

inline int blks(long n) { return (int)((n + 255) / 256); }

// C(MxN) = act(A(MxK) @ W(NxK)^T + bias); packs both operands to padded f16.
inline void gemm(const GemmCtx& g, const float* A, const float* W, const float* bias,
                 float* C, int M, int N, int K, int relu) {
  int Kp = (K + 31) & ~31;
  int Mp = (M + 15) & ~15;
  int Np = (N + 15) & ~15;
  k_pack_f16<<<blks((long)Mp * Kp), 256, 0, g.s>>>(A, g.Ah, M, K, Mp, Kp);
  k_pack_f16<<<blks((long)Np * Kp), 256, 0, g.s>>>(W, g.Wh, N, K, Np, Kp);
  dim3 b(32, 4);
  dim3 gr((unsigned)((Np / 16 + 3) / 4), (unsigned)(Mp / 16));
  k_gemm16<<<gr, b, 0, g.s>>>(g.Ah, g.Wh, bias, C, M, N, Kp, relu);
}

void run_mpnn(const GemmCtx& gc, Arena ar /*by value: scratch reclaimed*/,
              const float* nfeat, int nin, const float* efeat,
              const int* src, const int* dst, int N, int E, int H,
              const MPW& p, float* out, int decode) {
  hipStream_t s = gc.s;
  float* t0  = ar.f((size_t)N * H);
  float* hA  = ar.f((size_t)N * H);
  float* hB  = ar.f((size_t)N * H);
  float* g   = ar.f((size_t)E * EHID);
  float* Wp  = ar.f((size_t)EHID * H * H);
  float* Wq  = ar.f((size_t)H * H);
  float* deg = ar.f((size_t)N);
  float* P   = ar.f((size_t)N * EHID * H);
  float* Q   = ar.f((size_t)N * H);
  float* acc = ar.f((size_t)N * H);
  float* x   = ar.f((size_t)N * H);
  float* gi  = ar.f((size_t)N * 3 * H);
  float* gh  = ar.f((size_t)N * 3 * H);

  // project: h = lin(relu(lin(nfeat)))
  gemm(gc, nfeat, p.pw1, p.pb1, t0, N, H, nin, 1);
  gemm(gc, t0,    p.pw2, p.pb2, hA, N, H, H,   0);
  // edge MLP layer 1: g = relu(efeat @ ew1^T + eb1)
  gemm(gc, efeat, p.ew1, p.eb1, g, E, EHID, 4, 1);
  // reorder ew2/eb2 for node-level contraction
  {
    int tot = EHID * H * H;
    k_reorder_ew2<<<blks(tot), 256, 0, s>>>(p.ew2, p.eb2, Wp, Wq, H);
  }
  // in-degree
  k_zero<<<blks(N), 256, 0, s>>>(deg, N);
  k_deg<<<blks(E), 256, 0, s>>>(dst, deg, E);

  float* h  = hA;
  float* hn = hB;
  for (int step = 0; step < 2; ++step) {
    gemm(gc, h, Wp, nullptr, P, N, EHID * H, H, 0);   // P = h @ Wp^T
    gemm(gc, h, Wq, nullptr, Q, N, H,        H, 0);   // bias part of We
    long tot = (long)N * H;
    k_zero<<<blks(tot), 256, 0, s>>>(acc, tot);
    long et = (long)E * H;
    k_edge_msg_scatter<<<blks(et), 256, 0, s>>>(g, P, Q, src, dst, acc, E, H);
    k_agg_relu<<<blks(tot), 256, 0, s>>>(acc, deg, p.cb, x, N, H);
    gemm(gc, x, p.gwih, p.gbih, gi, N, 3 * H, H, 0);
    gemm(gc, h, p.gwhh, p.gbhh, gh, N, 3 * H, H, 0);
    k_gru<<<blks(tot), 256, 0, s>>>(gi, gh, h, hn, N, H);
    float* t = h; h = hn; hn = t;
  }
  if (decode) {
    gemm(gc, h,  p.dw1, p.db1, t0,  N, 32, H,  1);
    gemm(gc, t0, p.dw2, p.db2, out, N, 3,  32, 0);
  } else {
    hipMemcpyAsync(out, h, (size_t)N * H * sizeof(float),
                   hipMemcpyDeviceToDevice, s);
  }
}

} // namespace

extern "C" void kernel_launch(void* const* d_in, const int* in_sizes, int n_in,
                              void* d_out, int out_size, void* d_ws, size_t ws_size,
                              hipStream_t stream) {
  (void)in_sizes; (void)n_in; (void)out_size;
  const int R = 64;
  const int N1 = 6 * R * R;
  const int N3 = 6 * (R / 4) * (R / 4);
  const int N4 = 6 * (R / 8) * (R / 8);
  const int N5 = 6 * (R / 16) * (R / 16);
  const int E1 = 4 * N1, E3 = 4 * N3, E4 = 4 * N4, E5 = 4 * N5;

  int i = 0;
  const float* in_feat = (const float*)d_in[i++];
  const float* e1f = (const float*)d_in[i++];
  const float* e3f = (const float*)d_in[i++];
  const float* e4f = (const float*)d_in[i++];
  const float* e5f = (const float*)d_in[i++];
  const int* ei1 = (const int*)d_in[i++];
  const int* ei3 = (const int*)d_in[i++];
  const int* ei4 = (const int*)d_in[i++];
  const int* ei5 = (const int*)d_in[i++];

  // params in dict-insertion order (MP1..MP6 then up1w..up4b)
  MPW mp[6];
  for (int m = 0; m < 6; ++m) {
    MPW& q = mp[m];
    q.pw1 = (const float*)d_in[i++]; q.pb1 = (const float*)d_in[i++];
    q.pw2 = (const float*)d_in[i++]; q.pb2 = (const float*)d_in[i++];
    q.ew1 = (const float*)d_in[i++]; q.eb1 = (const float*)d_in[i++];
    q.ew2 = (const float*)d_in[i++]; q.eb2 = (const float*)d_in[i++];
    q.cb  = (const float*)d_in[i++];
    q.gwih = (const float*)d_in[i++]; q.gbih = (const float*)d_in[i++];
    q.gwhh = (const float*)d_in[i++]; q.gbhh = (const float*)d_in[i++];
    if (m == 5) {
      q.dw1 = (const float*)d_in[i++]; q.db1 = (const float*)d_in[i++];
      q.dw2 = (const float*)d_in[i++]; q.db2 = (const float*)d_in[i++];
    } else {
      q.dw1 = q.db1 = q.dw2 = q.db2 = nullptr;
    }
  }
  const float* up1w = (const float*)d_in[i++]; const float* up1b = (const float*)d_in[i++];
  const float* up2w = (const float*)d_in[i++]; const float* up2b = (const float*)d_in[i++];
  const float* up3w = (const float*)d_in[i++]; const float* up3b = (const float*)d_in[i++];
  const float* up4w = (const float*)d_in[i++]; const float* up4b = (const float*)d_in[i++];

  Arena ar{ (char*)d_ws, ws_size, 0 };

  // f16 GEMM staging (reused by every GEMM; stream-serialized so safe)
  GemmCtx gc;
  gc.s  = stream;
  gc.Ah = ar.h((size_t)4 * 1024 * 1024);   // >= max Mp*Kp (e1f pack: 98304*32)
  gc.Wh = ar.h((size_t)1 * 1024 * 1024);   // >= max Np*Kp (Wp MP3: 4096*128)

  // persistent intermediates
  float* h1   = ar.f((size_t)6 * 32 * 32 * 7);  // pool1 out
  float* h2i  = ar.f((size_t)N3 * 7);           // pool2 out (MP1 in)
  float* h2   = ar.f((size_t)N3 * 32);          // MP1 out (skip)
  float* h3i  = ar.f((size_t)N4 * 32);          // MP2 in
  float* h3   = ar.f((size_t)N4 * 64);          // MP2 out (skip)
  float* h4t  = ar.f((size_t)N4 * 16);          // scrambled-pool temp
  float* h4i  = ar.f((size_t)N5 * 64);          // MP3 in
  float* h4   = ar.f((size_t)N5 * 128);         // MP3 out
  float* u1   = ar.f((size_t)N4 * 128);
  float* c4   = ar.f((size_t)N4 * 192);
  float* m4   = ar.f((size_t)N4 * 98);
  float* u2   = ar.f((size_t)N3 * 98);
  float* c5   = ar.f((size_t)N3 * 130);
  float* m5   = ar.f((size_t)N3 * 60);
  float* u3   = ar.f((size_t)6 * 32 * 32 * 60);
  float* u4   = ar.f((size_t)N1 * 60);
  float* c6   = ar.f((size_t)N1 * 67);

  // ---- encoder ----
  k_pool2<<<blks(6L * 32 * 32 * 7), 256, 0, stream>>>(in_feat, h1, 64, 64, 7);
  k_pool2<<<blks((long)N3 * 7), 256, 0, stream>>>(h1, h2i, 32, 32, 7);
  run_mpnn(gc, ar, h2i, 7, e3f, ei3, ei3 + E3, N3, E3, 32, mp[0], h2, 0);

  k_pool2<<<blks((long)N4 * 32), 256, 0, stream>>>(h2, h3i, 16, 16, 32);
  run_mpnn(gc, ar, h3i, 32, e4f, ei4, ei4 + E4, N4, E4, 64, mp[1], h3, 0);

  // reference pools h3 with reshape(6,16,16,-1): 16 pseudo-channels + scramble
  k_pool2<<<blks((long)N4 * 16), 256, 0, stream>>>(h3, h4t, 16, 16, 16);
  k_scramble_h4<<<blks(96 * 64), 256, 0, stream>>>(h4t, h4i);
  run_mpnn(gc, ar, h4i, 64, e5f, ei5, ei5 + E5, N5, E5, 128, mp[2], h4, 0);

  // ---- decoder with skips ----
  k_convT<<<blks(6L * 8 * 8 * 128), 256, 0, stream>>>(h4, up1w, up1b, u1, 4, 4, 128, 128);
  k_concat<<<blks((long)N4 * 192), 256, 0, stream>>>(u1, h3, c4, N4, 128, 64);
  run_mpnn(gc, ar, c4, 192, e4f, ei4, ei4 + E4, N4, E4, 98, mp[3], m4, 0);

  k_convT<<<blks(6L * 16 * 16 * 98), 256, 0, stream>>>(m4, up2w, up2b, u2, 8, 8, 98, 98);
  k_concat<<<blks((long)N3 * 130), 256, 0, stream>>>(u2, h2, c5, N3, 98, 32);
  run_mpnn(gc, ar, c5, 130, e3f, ei3, ei3 + E3, N3, E3, 60, mp[4], m5, 0);

  k_convT<<<blks(6L * 32 * 32 * 60), 256, 0, stream>>>(m5, up3w, up3b, u3, 16, 16, 60, 60);
  k_convT<<<blks(6L * 64 * 64 * 60), 256, 0, stream>>>(u3, up4w, up4b, u4, 32, 32, 60, 60);
  k_concat<<<blks((long)N1 * 67), 256, 0, stream>>>(u4, in_feat, c6, N1, 60, 7);

  run_mpnn(gc, ar, c6, 67, e1f, ei1, ei1 + E1, N1, E1, 32, mp[5],
           (float*)d_out, 1);
}
